// UniLstmCellL1_69191923139140
// MI455X (gfx1250) — matrix operand order repaired and checked by reading, not compiled, not run on hardware
//
#include <hip/hip_runtime.h>

// ---- problem dims (compile-time, from the reference) -----------------------
#define T_STEPS 512
#define BATCH   512
#define IDIM    128
#define HDIM    256
#define KDIM    (IDIM + HDIM)   // 384: fused [x_t | h_t]
#define CLS     10

#define NWG_J   16              // 16 j-slices of 16 hidden columns
#define NWG_M   4               // 4 M-slices of 128 batch rows
#define NWG     (NWG_J * NWG_M) // 64 persistent workgroups
#define NTHR    256             // 8 wave32 per workgroup

typedef __attribute__((ext_vector_type(16))) _Float16 v16h;
typedef __attribute__((ext_vector_type(8)))  _Float16 v8h;
typedef __attribute__((ext_vector_type(4)))  _Float16 v4h;
typedef __attribute__((ext_vector_type(8)))  float    v8f;

__device__ __forceinline__ float sigmoidf_(float x) {
    return 1.0f / (1.0f + __expf(-x));
}
__device__ __forceinline__ float tanhf_(float x) {
    return 2.0f / (1.0f + __expf(-2.0f * x)) - 1.0f;
}

// ---- x (f32) -> xh (f16), one-time bandwidth pass --------------------------
__global__ void cvt_x_kernel(const float* __restrict__ x,
                             _Float16* __restrict__ xh, int n) {
    int i = (blockIdx.x * blockDim.x + threadIdx.x) * 4;
    if (i + 3 < n) {
        float4 v = *(const float4*)(x + i);
        v4h h;
        h[0] = (_Float16)v.x; h[1] = (_Float16)v.y;
        h[2] = (_Float16)v.z; h[3] = (_Float16)v.w;
        *(v4h*)(xh + i) = h;
    }
}

// ---- persistent fused LSTM recurrence --------------------------------------
__launch_bounds__(NTHR, 1)
__global__ void lstm_persistent_kernel(
    const float* __restrict__ W_ih,   // (1024, 128)
    const float* __restrict__ W_hh,   // (1024, 256)
    const float* __restrict__ b_ih,   // (1024,)
    const float* __restrict__ b_hh,   // (1024,)
    const int*   __restrict__ X_len,  // (512,)
    const _Float16* __restrict__ xh,  // (T, B, I) f16
    _Float16* __restrict__ hbuf,      // 2 x (B, H) f16, double buffered
    float* __restrict__ last_out,     // (T, H) f32 : h_t[X_len[t]-1, :]
    unsigned int* __restrict__ counter)
{
    // W slice for this j-slice: 64 rows (4 gates x 16 cols) x 384 K, f16 = 48 KB
    __shared__ _Float16 Wl[64 * KDIM];
    __shared__ float    bsl[64];

    const int tid  = threadIdx.x;
    const int wgj  = blockIdx.x & (NWG_J - 1);   // which 16-column slice of H
    const int wgm  = blockIdx.x >> 4;            // which 128-row slice of B
    const int wave = tid >> 5;
    const int lane = tid & 31;
    const int col  = lane & 15;                  // N index within tile
    const int half = lane >> 4;                  // lane-half selects K/M subrange
    const int jBase   = wgj * 16;
    const int rowBase = (wgm * 8 + wave) * 16;   // this wave's 16 batch rows

    // ---- stage weight slice into LDS (converted f32 -> f16), once ----------
    for (int idx = tid; idx < 64 * (KDIM / 8); idx += NTHR) {
        int row = idx / (KDIM / 8);              // 0..63 = gate*16 + rr
        int k   = (idx % (KDIM / 8)) * 8;
        int gt = row >> 4, rr = row & 15;
        int n  = gt * HDIM + jBase + rr;         // global gate-row index
        const float* src = (k < IDIM)
            ? (W_ih + (size_t)n * IDIM + k)
            : (W_hh + (size_t)n * HDIM + (k - IDIM));
        float4 a = *(const float4*)(src);
        float4 b = *(const float4*)(src + 4);
        _Float16* dst = &Wl[row * KDIM + k];
        dst[0] = (_Float16)a.x; dst[1] = (_Float16)a.y;
        dst[2] = (_Float16)a.z; dst[3] = (_Float16)a.w;
        dst[4] = (_Float16)b.x; dst[5] = (_Float16)b.y;
        dst[6] = (_Float16)b.z; dst[7] = (_Float16)b.w;
    }
    if (tid < 64) {
        int gt = tid >> 4, rr = tid & 15;
        int n  = gt * HDIM + jBase + rr;
        bsl[tid] = b_ih[n] + b_hh[n];
    }
    __syncthreads();

    // persistent cell state for this wave's 16x16 patch (f32 C-layout)
    v8f c_state;
    #pragma unroll
    for (int e = 0; e < 8; ++e) c_state[e] = 0.0f;

    const int m = rowBase + col;                 // A-frag row for this lane

    for (int t = 0; t < T_STEPS; ++t) {
        const _Float16* hr = hbuf + (size_t)(t & 1) * (BATCH * HDIM);
        _Float16*       hw = hbuf + (size_t)((t + 1) & 1) * (BATCH * HDIM);
        const int b_star = X_len[t] - 1;

        // ---- gates = [x_t | h_t] @ Wcat^T + bias, one 16x16 tile per gate --
        v8f acc[4];
        #pragma unroll
        for (int gt = 0; gt < 4; ++gt) {
            float bv = bsl[gt * 16 + col];       // bias depends only on column
            #pragma unroll
            for (int e = 0; e < 8; ++e) acc[gt][e] = bv;
        }

        #pragma unroll
        for (int kc = 0; kc < KDIM / 32; ++kc) {
            const int kb = kc * 32;
            // A fragment (16x32 f16): lane = row m, halves at K +8*half, +16+8*half
            const _Float16* arow = (kc < IDIM / 32)
                ? (xh + ((size_t)t * BATCH + m) * IDIM + kb)
                : (hr + (size_t)m * HDIM + (kb - IDIM));
            union { v16h v; v8h p[2]; } ua;
            ua.p[0] = *(const v8h*)(arow + 8 * half);
            ua.p[1] = *(const v8h*)(arow + 16 + 8 * half);
            #pragma unroll
            for (int gt = 0; gt < 4; ++gt) {
                // B fragment (32x16 f16): lane = column n, halves at K +8*half, +16+8*half
                const _Float16* wrow =
                    &Wl[(gt * 16 + col) * KDIM + kb + 8 * half];
                union { v16h v; v8h p[2]; } ub;
                ub.p[0] = *(const v8h*)(wrow);
                ub.p[1] = *(const v8h*)(wrow + 16);
                acc[gt] = __builtin_amdgcn_wmma_f32_16x16x32_f16(
                    false, ua.v, false, ub.v, (short)0, acc[gt], false, false);
            }
        }

        // ---- LSTM cell, elementwise on the f32 accumulator layout ----------
        float hn[8];
        #pragma unroll
        for (int e = 0; e < 8; ++e) {
            float ig = sigmoidf_(acc[0][e]);
            float fg = sigmoidf_(acc[1][e]);
            float gg = tanhf_(acc[2][e]);
            float og = sigmoidf_(acc[3][e]);
            float cn = fg * c_state[e] + ig * gg;
            c_state[e] = cn;
            hn[e] = og * tanhf_(cn);
        }

        // write h_new patch (rows m0..m0+7, col jBase+col) to alternate buffer
        const int m0 = rowBase + 8 * half;
        #pragma unroll
        for (int e = 0; e < 8; ++e)
            hw[(size_t)(m0 + e) * HDIM + jBase + col] = (_Float16)hn[e];

        // reference quirk: step t only needs h_t[X_len[t]-1, :]
        if ((b_star >> 3) == (m0 >> 3))
            last_out[(size_t)t * HDIM + jBase + col] = hn[b_star & 7];

        // ---- grid-wide step barrier (64 resident blocks) -------------------
        __threadfence();
        __syncthreads();
        if (tid == 0) {
            atomicAdd(counter, 1u);
            const unsigned target = (unsigned)NWG * (unsigned)(t + 1);
            while (__hip_atomic_load(counter, __ATOMIC_ACQUIRE,
                                     __HIP_MEMORY_SCOPE_AGENT) < target) {
                __builtin_amdgcn_s_sleep(1);
            }
        }
        __syncthreads();
        __threadfence();
    }
}

// ---- tiny FC(256->10) + softmax: one wave per output row -------------------
__global__ void fc_softmax_kernel(const float* __restrict__ last_out,
                                  const float* __restrict__ fc_w,
                                  const float* __restrict__ fc_b,
                                  float* __restrict__ out)
{
    const int r    = blockIdx.x;        // 0..511
    const int lane = threadIdx.x;       // 32 lanes
    const float* lo = last_out + (size_t)r * HDIM + lane * 8;
    float lv[8];
    #pragma unroll
    for (int q = 0; q < 8; ++q) lv[q] = lo[q];

    float lg[CLS];
    #pragma unroll
    for (int c = 0; c < CLS; ++c) {
        const float* wr = fc_w + (size_t)c * HDIM + lane * 8;
        float p = 0.0f;
        #pragma unroll
        for (int q = 0; q < 8; ++q) p += lv[q] * wr[q];
        #pragma unroll
        for (int off = 16; off > 0; off >>= 1) p += __shfl_xor(p, off, 32);
        lg[c] = p + fc_b[c];
    }
    float mx = lg[0];
    #pragma unroll
    for (int c = 1; c < CLS; ++c) mx = fmaxf(mx, lg[c]);
    float s = 0.0f;
    #pragma unroll
    for (int c = 0; c < CLS; ++c) s += __expf(lg[c] - mx);
    if (lane < CLS) out[(size_t)r * CLS + lane] = __expf(lg[lane] - mx) / s;
}

// ---- host-side launch ------------------------------------------------------
extern "C" void kernel_launch(void* const* d_in, const int* in_sizes, int n_in,
                              void* d_out, int out_size, void* d_ws, size_t ws_size,
                              hipStream_t stream) {
    (void)in_sizes; (void)n_in; (void)out_size; (void)ws_size;

    const float* x    = (const float*)d_in[0];
    const float* W_ih = (const float*)d_in[1];
    const float* W_hh = (const float*)d_in[2];
    const float* b_ih = (const float*)d_in[3];
    const float* b_hh = (const float*)d_in[4];
    const float* fc_w = (const float*)d_in[5];
    const float* fc_b = (const float*)d_in[6];
    const int*  X_len = (const int*)d_in[7];

    // workspace layout (256B aligned):
    //   xh       : T*B*I f16              = 64 MiB
    //   hbuf     : 2 * B*H f16            = 512 KiB (double buffered h)
    //   last_out : T*H f32                = 512 KiB
    //   counter  : grid-barrier counter
    char* ws = (char*)d_ws;
    size_t off = 0;
    auto take = [&](size_t bytes) -> char* {
        char* p = ws + off;
        off = (off + bytes + 255) & ~(size_t)255;
        return p;
    };
    _Float16* xh       = (_Float16*)take((size_t)T_STEPS * BATCH * IDIM * 2);
    _Float16* hbuf     = (_Float16*)take((size_t)2 * BATCH * HDIM * 2);
    float*    last_out = (float*)take((size_t)T_STEPS * HDIM * 4);
    unsigned* counter  = (unsigned*)take(256);

    // h0 = 0 (both parities for safety), barrier counter = 0 (every launch)
    (void)hipMemsetAsync(hbuf, 0, (size_t)2 * BATCH * HDIM * 2, stream);
    (void)hipMemsetAsync(counter, 0, 256, stream);

    const int nx = T_STEPS * BATCH * IDIM;
    cvt_x_kernel<<<(nx / 4 + NTHR - 1) / NTHR, NTHR, 0, stream>>>(x, xh, nx);

    lstm_persistent_kernel<<<NWG, NTHR, 0, stream>>>(
        W_ih, W_hh, b_ih, b_hh, X_len, xh, hbuf, last_out, counter);

    fc_softmax_kernel<<<T_STEPS, 32, 0, stream>>>(last_out, fc_w, fc_b,
                                                  (float*)d_out);
}
